// AgentAttention_13065290514875
// MI455X (gfx1250) — matrix operand order repaired and checked by reading, not compiled, not run on hardware
//
#include <hip/hip_runtime.h>
#include <hip/hip_bf16.h>

// ---------------------------------------------------------------------------
// AgentAttention for MI455X (gfx1250): all GEMMs via V_WMMA_F32_16X16X32_F16,
// attention operand staging via the Tensor Data Mover (tensor_load_to_lds).
// ---------------------------------------------------------------------------

typedef __attribute__((ext_vector_type(16))) _Float16 v16h;
typedef __attribute__((ext_vector_type(8)))  _Float16 v8h;
typedef __attribute__((ext_vector_type(4)))  _Float16 v4h;
typedef __attribute__((ext_vector_type(8)))  float    v8f;

#define BATCH 512
#define NTOK  196
#define CDIM  384
#define HEADS 12
#define DHEAD 32
#define AGENTS 49
#define WIMG  14
#define SCALE 0.17677669529663687f   // 32^-0.5

#if defined(__gfx1250__) && __has_builtin(__builtin_amdgcn_tensor_load_to_lds)
#define AA_USE_TDM 1
#else
#define AA_USE_TDM 0
#endif

// ------------------------- WMMA tile load helpers --------------------------
// A-matrix 16x32 f16: lanes 0-15 rows M=0..15 hold
// K = {kb..kb+15}, kb = (lane>>4)*16.  Two b128 loads per lane.
static __device__ __forceinline__ v16h load_A_f16(const _Float16* base, int stride, int lane) {
    int m  = lane & 15;
    int kb = (lane >> 4) * 16;
    const _Float16* p = base + m * stride + kb;
    v8h lo = *(const v8h*)(p);
    v8h hi = *(const v8h*)(p + 8);
    return __builtin_shufflevector(lo, hi, 0, 1, 2, 3, 4, 5, 6, 7,
                                   8, 9, 10, 11, 12, 13, 14, 15);
}

static __device__ __forceinline__ v16h load_A_f32(const float* base, int stride, int lane) {
    int m  = lane & 15;
    int kb = (lane >> 4) * 16;
    const float* p = base + m * stride + kb;
    v16h a;
#pragma unroll
    for (int j = 0; j < 8; ++j) {
        a[j]     = (_Float16)p[j];
        a[j + 8] = (_Float16)p[8 + j];
    }
    return a;
}

// B-matrix 32x16 f16 where the SOURCE stores B^T (src[n*stride + k]):
// lane = column n, K half by lane group -> one 32-byte contiguous load.
static __device__ __forceinline__ v16h load_Bt_f16(const _Float16* base, int stride, int lane) {
    int n  = lane & 15;
    int kb = (lane >> 4) * 16;
    return *(const v16h*)(base + n * stride + kb);
}

// B-matrix 32x16 f16 where the SOURCE stores B directly (src[k*stride + n]).
static __device__ __forceinline__ v16h load_B_f16(const _Float16* base, int stride, int lane) {
    int n  = lane & 15;
    int kb = (lane >> 4) * 16;
    const _Float16* p = base + kb * stride + n;
    v16h b;
#pragma unroll
    for (int j = 0; j < 16; ++j) b[j] = p[j * stride];
    return b;
}

static __device__ __forceinline__ float wave_max(float v) {
#pragma unroll
    for (int m = 16; m; m >>= 1) v = fmaxf(v, __shfl_xor(v, m, 32));
    return v;
}
static __device__ __forceinline__ float wave_sum(float v) {
#pragma unroll
    for (int m = 16; m; m >>= 1) v += __shfl_xor(v, m, 32);
    return v;
}

// ------------------------- Tensor Data Mover staging -----------------------
#if AA_USE_TDM
typedef __attribute__((ext_vector_type(4))) unsigned int u32x4;
typedef __attribute__((ext_vector_type(4))) int          i32x4;
typedef __attribute__((ext_vector_type(8))) int          i32x8;

// 2D tile load: 'rows' rows x 32 f16 elements, global row stride 1152 elems,
// packed contiguously into LDS at byte offset lds_off.
static __device__ __forceinline__ void tdm_load_2d(unsigned lds_off, const void* gptr, int rows) {
    unsigned long long ga = (unsigned long long)(size_t)gptr;
    u32x4 g0;
    g0[0] = 1u;                                            // count=1, user D#
    g0[1] = lds_off;                                       // lds_addr (bytes)
    g0[2] = (unsigned)(ga & 0xFFFFFFFFu);                  // global_addr[31:0]
    g0[3] = (unsigned)((ga >> 32) & 0x1FFFFFFu) | (2u << 30);  // [56:32] | type=2
    i32x8 g1;
    g1[0] = 1 << 16;                                       // data_size=1 (2 bytes)
    g1[1] = (int)((1152u & 0xFFFFu) << 16);                // tensor_dim0[15:0]
    g1[2] = (int)(((unsigned)rows & 0xFFFFu) << 16);       // dim0 hi=0 | tensor_dim1 lo
    g1[3] = (int)(32u << 16);                              // dim1 hi=0 | tile_dim0=32
    g1[4] = rows;                                          // tile_dim1 | tile_dim2=0
    g1[5] = 1152;                                          // tensor_dim0_stride lo32
    g1[6] = 0;                                             // stride0 hi | stride1 lo
    g1[7] = 0;
    i32x4 z4 = {0, 0, 0, 0};
#if defined(__clang_major__) && (__clang_major__ >= 23)
    i32x8 z8 = {0, 0, 0, 0, 0, 0, 0, 0};
    __builtin_amdgcn_tensor_load_to_lds(g0, g1, z4, z4, z8, 0);
#else
    __builtin_amdgcn_tensor_load_to_lds(g0, g1, z4, z4, 0);
#endif
}

static __device__ __forceinline__ void tdm_wait0() {
#if __has_builtin(__builtin_amdgcn_s_wait_tensorcnt)
    __builtin_amdgcn_s_wait_tensorcnt(0);
#else
    asm volatile("s_wait_tensorcnt 0x0" ::: "memory");
#endif
}
#endif  // AA_USE_TDM

// ------------------------- bias precompute ---------------------------------
static __device__ __forceinline__ float bilerp7(const float* s, int oy, int ox) {
    float cy = 0.5f * oy - 0.25f, cx = 0.5f * ox - 0.25f;
    int y0 = (int)floorf(cy), x0 = (int)floorf(cx);
    float fy = cy - (float)y0, fx = cx - (float)x0;
    int y0c = max(0, min(6, y0)),     x0c = max(0, min(6, x0));
    int y1c = max(0, min(6, y0 + 1)), x1c = max(0, min(6, x0 + 1));
    float v00 = s[y0c * 7 + x0c], v01 = s[y0c * 7 + x1c];
    float v10 = s[y1c * 7 + x0c], v11 = s[y1c * 7 + x1c];
    return v00 * (1.f - fy) * (1.f - fx) + v01 * (1.f - fy) * fx
         + v10 * fy * (1.f - fx)         + v11 * fy * fx;
}

__global__ void aa_bias_prep(const float* __restrict__ an_bias,
                             const float* __restrict__ ah_bias,
                             const float* __restrict__ aw_bias,
                             const float* __restrict__ na_bias,
                             const float* __restrict__ ha_bias,
                             const float* __restrict__ wa_bias,
                             float* __restrict__ biasA,   // [H][A][N]
                             float* __restrict__ biasQ)   // [H][N][A]
{
    int idx = blockIdx.x * blockDim.x + threadIdx.x;
    if (idx >= HEADS * AGENTS * NTOK) return;
    int n = idx % NTOK;
    int a = (idx / NTOK) % AGENTS;
    int h = idx / (NTOK * AGENTS);
    int y = n / WIMG, x = n % WIMG;

    float p1 = bilerp7(an_bias + (h * AGENTS + a) * 49, y, x);
    float p2 = ah_bias[(h * AGENTS + a) * WIMG + y] + aw_bias[(h * AGENTS + a) * WIMG + x];
    biasA[idx] = p1 + p2;

    float q1 = bilerp7(na_bias + (h * AGENTS + a) * 49, y, x);
    float q2 = ha_bias[(h * WIMG + y) * AGENTS + a] + wa_bias[(h * WIMG + x) * AGENTS + a];
    biasQ[(h * NTOK + n) * AGENTS + a] = q1 + q2;
}

// ------------------------- f32 -> f16 conversion (vector x4) ---------------
__global__ void aa_cvt_f16(const float* __restrict__ src, _Float16* __restrict__ dst, int n4) {
    int i = blockIdx.x * blockDim.x + threadIdx.x;
    if (i >= n4) return;
    float4 v = ((const float4*)src)[i];
    v4h h;
    h[0] = (_Float16)v.x; h[1] = (_Float16)v.y;
    h[2] = (_Float16)v.z; h[3] = (_Float16)v.w;
    ((v4h*)dst)[i] = h;
}

// ------------------------- QKV GEMM: (B*N,384) x (384,1152) ----------------
// one wave = one 16x16 tile; pure f16 b128 loads + wmma in the inner loop
__global__ void aa_gemm_qkv(const _Float16* __restrict__ xh,
                            const _Float16* __restrict__ wh,   // (1152,384) f16
                            _Float16* __restrict__ qkvb)       // (B*N,1152) f16
{
    const int lane = threadIdx.x & 31;
    const int wave = threadIdx.x >> 5;
    int tid = blockIdx.x * 8 + wave;
    const int NT = 1152 / 16;
    int tn = tid % NT, tm = tid / NT;
    int row0 = tm * 16, col0 = tn * 16;

    const _Float16* pa = xh + (size_t)(row0 + (lane & 15)) * CDIM + (lane >> 4) * 16;
    const _Float16* pb = wh + (size_t)(col0 + (lane & 15)) * CDIM + (lane >> 4) * 16;
    v8f acc = {};
#pragma unroll
    for (int k0 = 0; k0 < CDIM; k0 += 32) {
        v8h lo = *(const v8h*)(pa);
        v8h hi = *(const v8h*)(pa + 8);
        v16h a = __builtin_shufflevector(lo, hi, 0, 1, 2, 3, 4, 5, 6, 7,
                                         8, 9, 10, 11, 12, 13, 14, 15);
        v16h b = *(const v16h*)(pb);
        pa += 32; pb += 32;
        acc = __builtin_amdgcn_wmma_f32_16x16x32_f16(false, a, false, b, (short)0, acc, false, false);
    }
    const int hi4 = lane >> 4;
#pragma unroll
    for (int r = 0; r < 8; ++r) {
        int mm = r + 8 * hi4;
        qkvb[(size_t)(row0 + mm) * 1152 + col0 + (lane & 15)] = (_Float16)acc[r];
    }
}

// ------------------------- fused agent attention per (b,h) -----------------
// dynamic LDS (base offset 0, no static __shared__ in this kernel):
//  sQ  f16 208x32 @ 0       sK f16 208x32 @ 13312   sV f16 224x32 @ 26624
//  sAg f16  64x32 @ 40960   sAV f16 64x32 @ 45056   sS  f32 64x224|208x64 @ 49152
__global__ void aa_attn(const _Float16* __restrict__ qkvb,
                        const float* __restrict__ biasA,
                        const float* __restrict__ biasQ,
                        float* __restrict__ attn_out)     // (B*N,384) f32
{
    extern __shared__ char smem[];
    _Float16* sQ  = (_Float16*)(smem);
    _Float16* sK  = (_Float16*)(smem + 13312);
    _Float16* sV  = (_Float16*)(smem + 26624);
    _Float16* sAg = (_Float16*)(smem + 40960);
    _Float16* sAV = (_Float16*)(smem + 45056);
    float*    sS  = (float*)   (smem + 49152);

    const int b = blockIdx.x, h = blockIdx.y;
    const int tid  = threadIdx.x;
    const int lane = tid & 31;
    const int wave = tid >> 5;
    const size_t base = (size_t)(b * NTOK) * 1152 + h * DHEAD;

    // ---- stage Q,K,V head slices into LDS ----
#if AA_USE_TDM
    if (wave == 0) {
        tdm_load_2d(0u,     qkvb + base,       NTOK);   // Q
        tdm_load_2d(13312u, qkvb + base + 384, NTOK);   // K
        tdm_load_2d(26624u, qkvb + base + 768, NTOK);   // V
    }
    // zero the padded tail rows while the TDM streams
    for (int idx = tid; idx < (208 - NTOK) * 32; idx += 256) {
        sQ[NTOK * 32 + idx] = (_Float16)0.f;
        sK[NTOK * 32 + idx] = (_Float16)0.f;
    }
    for (int idx = tid; idx < (224 - NTOK) * 32; idx += 256)
        sV[NTOK * 32 + idx] = (_Float16)0.f;
    if (wave == 0) tdm_wait0();
    __syncthreads();
#else
    for (int idx = tid; idx < 208 * 32; idx += 256) {
        int n = idx >> 5, d = idx & 31;
        _Float16 qv = (_Float16)0.f, kv = (_Float16)0.f;
        if (n < NTOK) {
            qv = qkvb[base + (size_t)n * 1152 + d];
            kv = qkvb[base + (size_t)n * 1152 + 384 + d];
        }
        sQ[idx] = qv; sK[idx] = kv;
    }
    for (int idx = tid; idx < 224 * 32; idx += 256) {
        int n = idx >> 5, d = idx & 31;
        sV[idx] = (n < NTOK) ? qkvb[base + (size_t)n * 1152 + 768 + d] : (_Float16)0.f;
    }
    __syncthreads();
#endif

    // ---- agent pooling (2x2 mean of Q), pre-scaled by SCALE ----
    for (int idx = tid; idx < 64 * 32; idx += 256) {
        int a = idx >> 5, d = idx & 31;
        _Float16 val = (_Float16)0.f;
        if (a < AGENTS) {
            int y = (a / 7) * 2, x = (a % 7) * 2;
            float s = (float)sQ[(y * WIMG + x) * 32 + d]
                    + (float)sQ[(y * WIMG + x + 1) * 32 + d]
                    + (float)sQ[((y + 1) * WIMG + x) * 32 + d]
                    + (float)sQ[((y + 1) * WIMG + x + 1) * 32 + d];
            val = (_Float16)(s * 0.25f * SCALE);
        }
        sAg[idx] = val;
    }
    __syncthreads();

    // ---- S1 = (scale*agent) @ K^T : 64x208, K=32 ----
    for (int t = wave; t < 4 * 13; t += 8) {
        int tm = t % 4, tn = t / 4;
        v16h a = load_A_f16(sAg + tm * 16 * 32, 32, lane);
        v16h k = load_Bt_f16(sK + tn * 16 * 32, 32, lane);
        v8f  c = {};
        c = __builtin_amdgcn_wmma_f32_16x16x32_f16(false, a, false, k, (short)0, c, false, false);
        int hi = lane >> 4;
#pragma unroll
        for (int r = 0; r < 8; ++r)
            sS[(tm * 16 + r + 8 * hi) * 224 + tn * 16 + (lane & 15)] = c[r];
    }
    __syncthreads();

    // ---- softmax rows of S1 (+agent bias), zero pads ----
    for (int row = wave; row < 64; row += 8) {
        if (row < AGENTS) {
            float vals[7]; float mx = -1e30f;
#pragma unroll
            for (int i = 0; i < 7; ++i) {
                int col = lane + i * 32;
                float v = -1e30f;
                if (col < NTOK) v = sS[row * 224 + col] + biasA[(h * AGENTS + row) * NTOK + col];
                vals[i] = v; mx = fmaxf(mx, v);
            }
            mx = wave_max(mx);
            float sum = 0.f;
#pragma unroll
            for (int i = 0; i < 7; ++i) {
                int col = lane + i * 32;
                float e = (col < NTOK) ? __expf(vals[i] - mx) : 0.f;
                vals[i] = e; sum += e;
            }
            float inv = 1.f / wave_sum(sum);
#pragma unroll
            for (int i = 0; i < 7; ++i) {
                int col = lane + i * 32;
                sS[row * 224 + col] = (col < NTOK) ? vals[i] * inv : 0.f;
            }
        } else {
            for (int col = lane; col < 224; col += 32) sS[row * 224 + col] = 0.f;
        }
    }
    __syncthreads();

    // ---- agent_v = S1 @ V : 64x32, K=224 (7 WMMA steps) ----
    {
        int tm = wave >> 1, tn = wave & 1;          // exactly 8 tiles
        v8f c = {};
#pragma unroll
        for (int ks = 0; ks < 7; ++ks) {
            v16h a = load_A_f32(sS + tm * 16 * 224 + ks * 32, 224, lane);
            v16h v = load_B_f16(sV + ks * 32 * 32 + tn * 16, 32, lane);
            c = __builtin_amdgcn_wmma_f32_16x16x32_f16(false, a, false, v, (short)0, c, false, false);
        }
        int hi = lane >> 4;
#pragma unroll
        for (int r = 0; r < 8; ++r) {
            int mm = tm * 16 + r + 8 * hi;
            sAV[mm * 32 + tn * 16 + (lane & 15)] =
                (mm < AGENTS) ? (_Float16)c[r] : (_Float16)0.f;
        }
    }
    __syncthreads();

    // ---- S2 = Q @ (scale*agent)^T : 208x64, K=32 ----
    for (int t = wave; t < 13 * 4; t += 8) {
        int tm = t % 13, tn = t / 13;
        v16h a = load_A_f16(sQ + tm * 16 * 32, 32, lane);
        v16h g = load_Bt_f16(sAg + tn * 16 * 32, 32, lane);
        v8f  c = {};
        c = __builtin_amdgcn_wmma_f32_16x16x32_f16(false, a, false, g, (short)0, c, false, false);
        int hi = lane >> 4;
#pragma unroll
        for (int r = 0; r < 8; ++r)
            sS[(tm * 16 + r + 8 * hi) * 64 + tn * 16 + (lane & 15)] = c[r];
    }
    __syncthreads();

    // ---- softmax rows of S2 (+query bias), zero pads ----
    for (int row = wave; row < 208; row += 8) {
        if (row < NTOK) {
            int c0 = lane, c1 = lane + 32;
            float v0 = -1e30f, v1 = -1e30f;
            if (c0 < AGENTS) v0 = sS[row * 64 + c0] + biasQ[(h * NTOK + row) * AGENTS + c0];
            if (c1 < AGENTS) v1 = sS[row * 64 + c1] + biasQ[(h * NTOK + row) * AGENTS + c1];
            float mx = wave_max(fmaxf(v0, v1));
            float e0 = (c0 < AGENTS) ? __expf(v0 - mx) : 0.f;
            float e1 = (c1 < AGENTS) ? __expf(v1 - mx) : 0.f;
            float inv = 1.f / wave_sum(e0 + e1);
            sS[row * 64 + c0] = (c0 < AGENTS) ? e0 * inv : 0.f;
            sS[row * 64 + c1] = (c1 < AGENTS) ? e1 * inv : 0.f;
        } else {
            sS[row * 64 + lane] = 0.f;
            sS[row * 64 + lane + 32] = 0.f;
        }
    }
    __syncthreads();

    // ---- out = S2 @ agent_v : 208x32, K=64 (2 WMMA steps) ----
    for (int t = wave; t < 13 * 2; t += 8) {
        int tm = t % 13, tn = t / 13;
        v8f c = {};
#pragma unroll
        for (int ks = 0; ks < 2; ++ks) {
            v16h a = load_A_f32(sS + tm * 16 * 64 + ks * 32, 64, lane);
            v16h v = load_B_f16(sAV + ks * 32 * 32 + tn * 16, 32, lane);
            c = __builtin_amdgcn_wmma_f32_16x16x32_f16(false, a, false, v, (short)0, c, false, false);
        }
        int hi = lane >> 4;
#pragma unroll
        for (int r = 0; r < 8; ++r) {
            int token = tm * 16 + r + 8 * hi;
            if (token < NTOK)
                attn_out[(size_t)(b * NTOK + token) * CDIM + h * DHEAD + tn * 16 + (lane & 15)] = c[r];
        }
    }
}

// ------------------------- depthwise 3x3 + residual ------------------------
__global__ void aa_dwc_add(const _Float16* __restrict__ qkvb,
                           const float* __restrict__ attn_out,
                           const float* __restrict__ dwc_w,   // (C,1,3,3)
                           const float* __restrict__ dwc_b,
                           _Float16* __restrict__ tb)         // (B*N,C) f16
{
    int idx = blockIdx.x * blockDim.x + threadIdx.x;
    const int total = BATCH * NTOK * CDIM;
    if (idx >= total) return;
    int c = idx % CDIM;
    int n = (idx / CDIM) % NTOK;
    int b = idx / (CDIM * NTOK);
    int y = n / WIMG, x = n % WIMG;

    float acc = dwc_b[c];
#pragma unroll
    for (int ky = 0; ky < 3; ++ky) {
        int yy = y + ky - 1;
        if (yy < 0 || yy >= WIMG) continue;
#pragma unroll
        for (int kx = 0; kx < 3; ++kx) {
            int xx = x + kx - 1;
            if (xx < 0 || xx >= WIMG) continue;
            acc += dwc_w[c * 9 + ky * 3 + kx] *
                   (float)qkvb[(size_t)(b * NTOK + yy * WIMG + xx) * 1152 + 768 + c];
        }
    }
    tb[idx] = (_Float16)(acc + attn_out[idx]);
}

// ------------------------- output projection GEMM --------------------------
__global__ void aa_gemm_proj(const _Float16* __restrict__ tb,
                             const _Float16* __restrict__ wh,  // (384,384) f16
                             const float* __restrict__ proj_b,
                             float* __restrict__ out)
{
    const int lane = threadIdx.x & 31;
    const int wave = threadIdx.x >> 5;
    int tid = blockIdx.x * 8 + wave;
    const int NT = CDIM / 16;
    int tn = tid % NT, tm = tid / NT;
    int row0 = tm * 16, col0 = tn * 16;

    const _Float16* pa = tb + (size_t)(row0 + (lane & 15)) * CDIM + (lane >> 4) * 16;
    const _Float16* pb = wh + (size_t)(col0 + (lane & 15)) * CDIM + (lane >> 4) * 16;
    v8f acc = {};
#pragma unroll
    for (int k0 = 0; k0 < CDIM; k0 += 32) {
        v8h lo = *(const v8h*)(pa);
        v8h hi = *(const v8h*)(pa + 8);
        v16h a = __builtin_shufflevector(lo, hi, 0, 1, 2, 3, 4, 5, 6, 7,
                                         8, 9, 10, 11, 12, 13, 14, 15);
        v16h b = *(const v16h*)(pb);
        pa += 32; pb += 32;
        acc = __builtin_amdgcn_wmma_f32_16x16x32_f16(false, a, false, b, (short)0, acc, false, false);
    }
    const int hi4 = lane >> 4;
#pragma unroll
    for (int r = 0; r < 8; ++r) {
        int mm = r + 8 * hi4;
        int cc = col0 + (lane & 15);
        out[(size_t)(row0 + mm) * CDIM + cc] = acc[r] + proj_b[cc];
    }
}

// ---------------------------------------------------------------------------
extern "C" void kernel_launch(void* const* d_in, const int* in_sizes, int n_in,
                              void* d_out, int out_size, void* d_ws, size_t ws_size,
                              hipStream_t stream) {
    const float* x       = (const float*)d_in[0];
    const float* qkv_w   = (const float*)d_in[1];
    const float* proj_w  = (const float*)d_in[2];
    const float* proj_b  = (const float*)d_in[3];
    const float* dwc_w   = (const float*)d_in[4];
    const float* dwc_b   = (const float*)d_in[5];
    const float* an_bias = (const float*)d_in[6];
    const float* ah_bias = (const float*)d_in[7];
    const float* aw_bias = (const float*)d_in[8];
    const float* na_bias = (const float*)d_in[9];
    const float* ha_bias = (const float*)d_in[10];
    const float* wa_bias = (const float*)d_in[11];
    float* out = (float*)d_out;
    (void)in_sizes; (void)n_in; (void)out_size; (void)ws_size;

    char* ws = (char*)d_ws;
    const size_t BN = (size_t)BATCH * NTOK;
    size_t off = 0;
    _Float16* qkvb  = (_Float16*)(ws + off); off += BN * 1152 * 2;   // 231 MB
    float*    attn  = (float*)   (ws + off); off += BN * 384 * 4;    // 154 MB
    _Float16* tb    = (_Float16*)(ws + off); off += BN * 384 * 2;    //  77 MB
    _Float16* xh    = (_Float16*)(ws + off); off += BN * 384 * 2;    //  77 MB
    _Float16* qkvwh = (_Float16*)(ws + off); off += (size_t)1152 * 384 * 2;
    _Float16* projwh= (_Float16*)(ws + off); off += (size_t)384 * 384 * 2;
    float*    biasA = (float*)   (ws + off); off += (size_t)HEADS * AGENTS * NTOK * 4;
    float*    biasQ = (float*)   (ws + off);

    // 1) bias tables
    {
        int total = HEADS * AGENTS * NTOK;
        aa_bias_prep<<<(total + 255) / 256, 256, 0, stream>>>(
            an_bias, ah_bias, aw_bias, na_bias, ha_bias, wa_bias, biasA, biasQ);
    }
    // 2) one-shot f32 -> f16 of x and both weight matrices
    {
        int n4 = (int)(BN * 384 / 4);
        aa_cvt_f16<<<(n4 + 255) / 256, 256, 0, stream>>>(x, xh, n4);
        n4 = 1152 * 384 / 4;
        aa_cvt_f16<<<(n4 + 255) / 256, 256, 0, stream>>>(qkv_w, qkvwh, n4);
        n4 = 384 * 384 / 4;
        aa_cvt_f16<<<(n4 + 255) / 256, 256, 0, stream>>>(proj_w, projwh, n4);
    }
    // 3) QKV GEMM: 6272*72 tiles, 8 waves/block
    {
        int tiles = (BATCH * NTOK / 16) * (1152 / 16);
        aa_gemm_qkv<<<tiles / 8, 256, 0, stream>>>(xh, qkvwh, qkvb);
    }
    // 4) fused attention, one block per (b,h); ~104 KB dynamic LDS (CDNA5 WGP)
    {
        dim3 grid(BATCH, HEADS);
        size_t lds = 49152 + 64 * 224 * sizeof(float);   // 106,496 B
        aa_attn<<<grid, 256, lds, stream>>>(qkvb, biasA, biasQ, attn);
    }
    // 5) depthwise conv + residual
    {
        int total = BATCH * NTOK * CDIM;
        aa_dwc_add<<<(total + 255) / 256, 256, 0, stream>>>(qkvb, attn, dwc_w, dwc_b, tb);
    }
    // 6) projection GEMM: 6272*24 tiles
    {
        int tiles = (BATCH * NTOK / 16) * (CDIM / 16);
        aa_gemm_proj<<<tiles / 8, 256, 0, stream>>>(tb, projwh, proj_b, out);
    }
}
